// GDN_57956288692310
// MI455X (gfx1250) — matrix operand order repaired and checked by reading, not compiled, not run on hardware
//
#include <hip/hip_runtime.h>

typedef __attribute__((ext_vector_type(16))) _Float16 v16h;
typedef __attribute__((ext_vector_type(8)))  float    v8f;
typedef __attribute__((ext_vector_type(4)))  float    v4f;

#define CH   128      // channels (M and K)
#define HW   16384    // pixels per batch image (128*128)
#define NT   128      // pixels per workgroup tile
#define NKK  4        // K chunks of 32
#define NNT  8        // 16-pixel subtiles per tile

// GDN: norm = gamma @ (x*x) + beta  (per-pixel channel mixing), out = x * rsqrt(norm)
// f16 hi/lo split-compensated WMMA (3 products) for ~fp32 accuracy at f16 matrix rates.
__global__ __launch_bounds__(256)
void gdn_wmma_kernel(const float* __restrict__ x,
                     const float* __restrict__ gamma,
                     const float* __restrict__ beta,
                     float* __restrict__ out) {
  // B-fragment-ordered LDS: [kk(4)][ntile(8)][lane(32)][16 halves]
  __shared__ _Float16 sHi[NKK * NNT * 32 * 16];   // 32 KB
  __shared__ _Float16 sLo[NKK * NNT * 32 * 16];   // 32 KB

  const int tid  = threadIdx.x;
  const int lane = tid & 31;
  const int w    = tid >> 5;                 // wave id == M-tile (output-channel tile)
  const int b    = blockIdx.x >> 7;          // batch
  const int p0   = (blockIdx.x & 127) * NT;  // pixel-tile base within batch image
  const float* xb = x   + (size_t)b * CH * HW + p0;
  float*       ob = out + (size_t)b * CH * HW + p0;

  // ---------- Phase 0: gamma A-fragments (hi/lo f16) in registers ----------
  // A 16x32 f16 k order used here: lanes 0-15 (M=0-15): VGPR0-7 = K 0..15 (contiguous)
  //                     lanes 16-31:         VGPR0-7 = K 16..31
  v16h aHi[NKK], aLo[NKK];
  {
    const int m     = w * 16 + (lane & 15);
    const int khalf = lane >> 4;
    const float* grow = gamma + m * CH + khalf * 16;
#pragma unroll
    for (int kk = 0; kk < NKK; ++kk) {
      const float* g0 = grow + kk * 32;      // halves 0..7  (K = kk*32 + khalf*16 + h)
      const float* g1 = g0 + 8;              // halves 8..15 (K = +8)
#pragma unroll
      for (int h = 0; h < 8; ++h) {
        float f0 = g0[h], f1 = g1[h];
        _Float16 h0 = (_Float16)f0, h1 = (_Float16)f1;
        aHi[kk][h]     = h0;
        aHi[kk][8 + h] = h1;
        aLo[kk][h]     = (_Float16)(f0 - (float)h0);
        aLo[kk][8 + h] = (_Float16)(f1 - (float)h1);
      }
    }
  }

  // ---------- Phase 1: x tile -> s = x*x -> f16 hi/lo -> LDS (B-frag order) ----------
  // B 32x16 f16 layout: lanes 0-15 (N=0-15) hold K 0..15 (2/VGPR), lanes 16-31 hold K 16..31.
#pragma unroll
  for (int i = 0; i < 16; ++i) {
    const int e4 = i * 256 + tid;            // float4 index in 128x128 tile
    const int e  = e4 << 2;
    const int k  = e >> 7;                   // channel row
    const int n  = e & 127;                  // pixel col (multiple of 4)
    v4f xv = *(const v4f*)(xb + (size_t)k * HW + n);
    const int kk    = k >> 5;
    const int kr    = k & 31;
    const int lane0 = (n & 15) + ((kr >> 4) << 4);
    const int base  = ((kk << 3) + (n >> 4)) * 512 + (kr & 15);
#pragma unroll
    for (int j = 0; j < 4; ++j) {
      float s = xv[j] * xv[j];
      _Float16 hi = (_Float16)s;
      float lo = s - (float)hi;
      const int off = base + (lane0 + j) * 16;
      sHi[off] = hi;
      sLo[off] = (_Float16)lo;
    }
  }
  __syncthreads();

  // ---------- Phase 2: WMMA GEMM + epilogue ----------
  // C/D 16x16 f32 layout: VGPR r -> (M = r + 8*(lane>>4), N = lane&15)
  const int obase = w * 16 + ((lane >> 4) << 3);
  float betaReg[8];
#pragma unroll
  for (int r = 0; r < 8; ++r) betaReg[r] = beta[obase + r];

  for (int nt = 0; nt < NNT; ++nt) {
    v8f acc = {};
#pragma unroll
    for (int kk = 0; kk < NKK; ++kk) {
      const int fb = (kk * NNT + nt) * 512 + lane * 16;
      v16h bHi = *(const v16h*)(&sHi[fb]);
      v16h bLo = *(const v16h*)(&sLo[fb]);
      // norm += gHi*sHi + gHi*sLo + gLo*sHi   (drop lo*lo, ~2^-22 relative)
      acc = __builtin_amdgcn_wmma_f32_16x16x32_f16(false, aHi[kk], false, bHi,
                                                   (short)0, acc, false, false);
      acc = __builtin_amdgcn_wmma_f32_16x16x32_f16(false, aHi[kk], false, bLo,
                                                   (short)0, acc, false, false);
      acc = __builtin_amdgcn_wmma_f32_16x16x32_f16(false, aLo[kk], false, bHi,
                                                   (short)0, acc, false, false);
    }
    const int p = nt * 16 + (lane & 15);
#pragma unroll
    for (int r = 0; r < 8; ++r) {
      const size_t idx = (size_t)(obase + r) * HW + p;
      float xv   = xb[idx];                  // L2 hit: tile was just streamed
      float norm = acc[r] + betaReg[r];
      // norm >= beta ~= 1.0: no denorm/overflow risk, use raw v_rsq_f32 + one
      // Newton step instead of libm rsqrtf's guarded (v_cmp + rescale) path.
      float rs = __builtin_amdgcn_rsqf(norm);
      rs = rs * (1.5f - 0.5f * norm * rs * rs);   // ~fp32-accurate rsqrt
      ob[idx] = xv * rs;
    }
  }
}

extern "C" void kernel_launch(void* const* d_in, const int* in_sizes, int n_in,
                              void* d_out, int out_size, void* d_ws, size_t ws_size,
                              hipStream_t stream) {
  const float* x     = (const float*)d_in[0];
  const float* gamma = (const float*)d_in[1];
  const float* beta  = (const float*)d_in[2];
  float* out = (float*)d_out;
  // one WG per (batch, 128-pixel) tile: total_elems / (CH * NT)
  const int blocks = in_sizes[0] / (CH * NT);
  gdn_wmma_kernel<<<dim3(blocks), dim3(256), 0, stream>>>(x, gamma, beta, out);
}
